// RoomTransformerModel_23708219474779
// MI455X (gfx1250) — matrix operand order repaired and checked by reading, not compiled, not run on hardware
//
#include <hip/hip_runtime.h>
#include <hip/hip_bf16.h>

// ---------------------------------------------------------------------------
// RoomTransformerModel on MI455X (gfx1250): f16 WMMA + async LDS staging.
// ---------------------------------------------------------------------------

#define B_    32
#define S_    513           // 1 global row + 512 rooms
#define SP    544           // padded seq (multiple of 32); 34 tiles of 16
#define T_    (B_ * SP)     // 17408 padded rows total (multiple of 128)
#define E_    512
#define H_    8
#define KW_   64
#define HID_  2048
#define L_    4
#define NOUT_ 256

typedef _Float16 half_t;
typedef __attribute__((ext_vector_type(16))) _Float16 v16h;
typedef __attribute__((ext_vector_type(8)))  float    v8f;

union HF16 {
    v16h v;
    half_t h[16];
    uint4 q[2];
};

__device__ inline float gelu_exact(float x) {
    return 0.5f * x * (1.0f + erff(x * 0.70710678118654752440f));
}

__device__ inline v8f wmma_f16(const HF16& a, const HF16& b, v8f c) {
    // D = A(16x32 f16) * B(32x16 f16) + C(16x16 f32)
    return __builtin_amdgcn_wmma_f32_16x16x32_f16(
        /*neg_a=*/false, a.v, /*neg_b=*/false, b.v,
        /*c_mod=*/(short)0, c, /*reuse_a=*/false, /*reuse_b=*/false);
}

// ---- CDNA5 async global->LDS copy (ASYNCcnt), inline asm per ISA 15.18.3 ----
__device__ inline void async_copy_b128(void* lds_dst, const void* gsrc) {
    // LDS byte address = low 32 bits of the generic pointer (ISA 10.2 aperture)
    unsigned int lds = (unsigned int)(unsigned long long)lds_dst;
    asm volatile("global_load_async_to_lds_b128 %0, %1, off"
                 :: "v"(lds), "v"(gsrc)
                 : "memory");
}
__device__ inline void wait_asynccnt0() {
    asm volatile("s_wait_asynccnt 0" ::: "memory");
}

// ---------------------------------------------------------------------------
// f32 -> f16 conversion (weights)
// ---------------------------------------------------------------------------
__global__ void f32_to_f16_kernel(const float* __restrict__ src,
                                  half_t* __restrict__ dst, int n) {
    int i = blockIdx.x * blockDim.x + threadIdx.x;
    if (i < n) dst[i] = (half_t)src[i];
}

// ---------------------------------------------------------------------------
// Embedding assembly: builds X (f32 + f16), zero pad rows.
// grid = T_, block = 128 (each thread does 4 consecutive e)
// ---------------------------------------------------------------------------
__global__ __launch_bounds__(128) void embed_kernel(
    const unsigned char* __restrict__ room_mask,   // (32,512) bool
    const int* __restrict__ rpx, const int* __restrict__ rpy,     // (32,512)
    const int* __restrict__ map_door_id,                           // (32)
    const float* __restrict__ steps, const float* __restrict__ temp,
    const float* __restrict__ mc,
    const int* __restrict__ room_dir,                              // (512,3)
    const int* __restrict__ rhx, const int* __restrict__ rhy,      // (512)
    const float* __restrict__ Wg,   // (515,512)
    const float* __restrict__ bg,   // (512)
    const float* __restrict__ pex, const float* __restrict__ pey,  // (72,512)
    const float* __restrict__ rem, const float* __restrict__ unp,  // (512,512)
    const float* __restrict__ mde,                                  // (512,512)
    const float* __restrict__ mpx, const float* __restrict__ mpy,   // (72,512)
    float* __restrict__ Xf, half_t* __restrict__ Xh)
{
    const int idx = blockIdx.x;
    const int b = idx / SP, s = idx % SP;
    const size_t base = (size_t)idx * E_;
    const int e0 = threadIdx.x * 4;
    float v[4];
    if (s >= S_) {
        v[0] = v[1] = v[2] = v[3] = 0.0f;
    } else if (s == 0) {
        const int door = map_door_id[b];
        const int rid  = room_dir[door * 3 + 0];
        const int px   = rpx[b * 512 + rid] + room_dir[door * 3 + 1];
        const int py   = rpy[b * 512 + rid] + room_dir[door * 3 + 2];
        const float c0 = steps[b] * (1.0f / 512.0f);
        const float c1 = logf(temp[b]);
        const float c2 = mc[b];
        #pragma unroll
        for (int j = 0; j < 4; j++) {
            const int e = e0 + j;
            float acc = bg[e] + mde[(size_t)door * E_ + e]
                      + mpx[(size_t)px * E_ + e] + mpy[(size_t)py * E_ + e];
            acc += c0 * Wg[(size_t)512 * E_ + e]
                 + c1 * Wg[(size_t)513 * E_ + e]
                 + c2 * Wg[(size_t)514 * E_ + e];
            v[j] = acc;
        }
        for (int r = 0; r < 512; r++) {
            if (room_mask[b * 512 + r]) {
                #pragma unroll
                for (int j = 0; j < 4; j++) v[j] += Wg[(size_t)r * E_ + e0 + j];
            }
        }
    } else {
        const int r = s - 1;
        if (room_mask[b * 512 + r]) {
            const int ax = rpx[b * 512 + r] + rhx[r];
            const int ay = rpy[b * 512 + r] + rhy[r];
            #pragma unroll
            for (int j = 0; j < 4; j++) {
                const int e = e0 + j;
                v[j] = pex[(size_t)ax * E_ + e] + pey[(size_t)ay * E_ + e]
                     + rem[(size_t)r * E_ + e];
            }
        } else {
            #pragma unroll
            for (int j = 0; j < 4; j++) v[j] = unp[(size_t)r * E_ + e0 + j];
        }
    }
    #pragma unroll
    for (int j = 0; j < 4; j++) {
        Xf[base + e0 + j] = v[j];
        Xh[base + e0 + j] = (half_t)v[j];
    }
}

// ---------------------------------------------------------------------------
// Generic f16 WMMA GEMM: C(MxN) = epi(A(MxK) @ B(KxN))
// block = 256 threads (8 waves); block tile 128xBN (BN = 64 or 128);
// wave tile 32x(BN/2); kstep 32; double-buffered LDS; A staged with
// global_load_async_to_lds_b128 (ASYNCcnt), B transposed into LDS.
// EPI: 0 = store f16; 1 = gelu -> f16; 2 = residual: f32 += acc, store f32+f16
// grid = (N/BN, M/128)
// ---------------------------------------------------------------------------
template <int EPI, int BN>
__global__ __launch_bounds__(256) void gemm_kernel(
    const half_t* __restrict__ A, const half_t* __restrict__ Bw,
    int K, int N, half_t* __restrict__ Ch, float* __restrict__ Resid)
{
    constexpr int NS = BN / 32;     // n-subtiles per wave (2 or 4)
    constexpr int WN = BN / 2;      // wave n-extent
    __shared__ half_t Al[2][128 * 40];  // row-major A tiles, stride 40 halves
    __shared__ half_t Bl[2][BN * 40];   // B tiles transposed: Bl[buf][n][k]

    const int tid  = threadIdx.x;
    const int lane = tid & 31;
    const int wave = tid >> 5;
    const int lm = lane & 15, lh = lane >> 4;
    const int m_blk = blockIdx.y * 128;
    const int n_blk = blockIdx.x * BN;
    const int wm = wave >> 1;   // 0..3  (M sub-block)
    const int wn = wave & 1;    // 0..1  (N sub-block)

    v8f c[2][NS];
    #pragma unroll
    for (int i = 0; i < 2; i++)
        #pragma unroll
        for (int j = 0; j < NS; j++)
            c[i][j] = (v8f){0.f, 0.f, 0.f, 0.f, 0.f, 0.f, 0.f, 0.f};

    const int arow = tid >> 1;                 // 0..127
    const int aseg = (tid & 1) * 16;           // 0 / 16 halves
    const int brow = tid >> 3;                 // 0..31  (k within tile)
    const int bseg = (tid & 7) * (BN / 8);     // n within tile (halves)

    auto stageA = [&](int k0, int buf) {
        const half_t* gp = A + (size_t)(m_blk + arow) * K + k0 + aseg;
        async_copy_b128(&Al[buf][arow * 40 + aseg], gp);
        async_copy_b128(&Al[buf][arow * 40 + aseg + 8], gp + 8);
    };
    auto stageB = [&](int k0, int buf) {
        const half_t* gp = Bw + (size_t)(k0 + brow) * N + n_blk + bseg;
        #pragma unroll
        for (int s8 = 0; s8 < BN / 64; s8++) {
            uint4 bv = *(const uint4*)(gp + s8 * 8);
            const half_t* bh = (const half_t*)&bv;
            #pragma unroll
            for (int j = 0; j < 8; j++)
                Bl[buf][(bseg + s8 * 8 + j) * 40 + brow] = bh[j];
        }
    };

    // prologue: stage tile 0
    stageA(0, 0);
    stageB(0, 0);
    wait_asynccnt0();
    __syncthreads();

    int buf = 0;
    for (int k0 = 0; k0 < K; k0 += 32) {
        // stage next tile into the other buffer while computing this one
        if (k0 + 32 < K) {
            stageA(k0 + 32, buf ^ 1);
            __builtin_prefetch(Bw + (size_t)(k0 + 64 + brow) * N + n_blk + bseg, 0, 0);
            stageB(k0 + 32, buf ^ 1);
        }

        // ---- fragments per ISA 7.12.2 layouts (contiguous 16B LDS loads)
        HF16 af[2], bf[NS];
        #pragma unroll
        for (int ms = 0; ms < 2; ms++) {
            const half_t* p = &Al[buf][(wm * 32 + ms * 16 + lm) * 40 + 8 * lh];
            af[ms].q[0] = *(const uint4*)p;
            af[ms].q[1] = *(const uint4*)(p + 16);
        }
        #pragma unroll
        for (int ns = 0; ns < NS; ns++) {
            const half_t* p = &Bl[buf][(wn * WN + ns * 16 + lm) * 40 + 8 * lh];
            bf[ns].q[0] = *(const uint4*)p;
            bf[ns].q[1] = *(const uint4*)(p + 16);
        }
        #pragma unroll
        for (int ms = 0; ms < 2; ms++)
            #pragma unroll
            for (int ns = 0; ns < NS; ns++)
                c[ms][ns] = wmma_f16(af[ms], bf[ns], c[ms][ns]);

        wait_asynccnt0();   // next tile's async A writes have landed
        __syncthreads();
        buf ^= 1;
    }

    // ---- epilogue
    #pragma unroll
    for (int ms = 0; ms < 2; ms++) {
        #pragma unroll
        for (int ns = 0; ns < NS; ns++) {
            const int col = n_blk + wn * WN + ns * 16 + lm;
            #pragma unroll
            for (int v = 0; v < 8; v++) {
                const int row = m_blk + wm * 32 + ms * 16 + v + 8 * lh;
                const size_t off = (size_t)row * N + col;
                const float val = c[ms][ns][v];
                if (EPI == 0) {
                    Ch[off] = (half_t)val;
                } else if (EPI == 1) {
                    Ch[off] = (half_t)gelu_exact(val);
                } else {
                    const float r = Resid[off] + val;
                    Resid[off] = r;
                    Ch[off] = (half_t)r;
                }
            }
        }
    }
}

// ---------------------------------------------------------------------------
// V transpose: Vh (T_ x 64) -> Vt (64 x T_) so P@V B-fragments are contiguous
// ---------------------------------------------------------------------------
__global__ void transpose_v_kernel(const half_t* __restrict__ Vh,
                                   half_t* __restrict__ Vt) {
    int idx = blockIdx.x * blockDim.x + threadIdx.x;
    if (idx >= T_ * 64) return;
    int t = idx >> 6, w = idx & 63;
    Vt[(size_t)w * T_ + t] = Vh[idx];
}

// ---------------------------------------------------------------------------
// Flash attention: 1 wave per (batch, head, 16-query tile).
// scores = Q(16x64)@K^T, online softmax over t (mask t>=513), A += P@V.
// Epilogue: G = f16(gelu(A / l)) written at [row, head*64 + w].
// grid = B_*H_*(SP/16), block = 32
// ---------------------------------------------------------------------------
__global__ __launch_bounds__(32) void attn_kernel(
    const half_t* __restrict__ Q,    // (T_, 512) row-major, head-major cols
    const half_t* __restrict__ Kk,   // (T_, 64)  shared across heads
    const half_t* __restrict__ Vt,   // (64, T_)  transposed V
    half_t* __restrict__ G)          // (T_, 512)
{
    __shared__ half_t Pl[16 * 32];   // probability tile, A-frag staging

    const int lane = threadIdx.x;
    const int lm = lane & 15, lh = lane >> 4;
    const int idx = blockIdx.x;
    const int mt = idx % (SP / 16);
    const int h  = (idx / (SP / 16)) % H_;
    const int b  = idx / ((SP / 16) * H_);
    const int qrow = b * SP + mt * 16 + lm;

    HF16 qf[2];
    #pragma unroll
    for (int ks = 0; ks < 2; ks++) {
        const half_t* p = Q + (size_t)qrow * E_ + h * 64 + ks * 32 + 8 * lh;
        qf[ks].q[0] = *(const uint4*)p;
        qf[ks].q[1] = *(const uint4*)(p + 16);
    }

    v8f oacc[4];
    #pragma unroll
    for (int nt = 0; nt < 4; nt++)
        oacc[nt] = (v8f){0.f, 0.f, 0.f, 0.f, 0.f, 0.f, 0.f, 0.f};
    float rmax[8], rsum[8];
    #pragma unroll
    for (int v = 0; v < 8; v++) { rmax[v] = -1e30f; rsum[v] = 0.0f; }

    for (int tc = 0; tc < SP / 32; tc++) {
        const int t0 = tc * 32;
        float s[2][8];
        #pragma unroll
        for (int st = 0; st < 2; st++) {
            const int trow = b * SP + t0 + st * 16 + lm;
            v8f acc = (v8f){0.f, 0.f, 0.f, 0.f, 0.f, 0.f, 0.f, 0.f};
            #pragma unroll
            for (int ks = 0; ks < 2; ks++) {
                HF16 kf;
                const half_t* p = Kk + (size_t)trow * 64 + ks * 32 + 8 * lh;
                kf.q[0] = *(const uint4*)p;
                kf.q[1] = *(const uint4*)(p + 16);
                acc = wmma_f16(qf[ks], kf, acc);
            }
            const bool valid = (t0 + st * 16 + lm) < S_;
            #pragma unroll
            for (int v = 0; v < 8; v++)
                s[st][v] = valid ? acc[v] * 0.125f : -1e30f;
        }

        // ---- online softmax update (row reductions within 16-lane halves)
        float pch[2][8];
        #pragma unroll
        for (int v = 0; v < 8; v++) {
            float cm = fmaxf(s[0][v], s[1][v]);
            #pragma unroll
            for (int off = 1; off < 16; off <<= 1)
                cm = fmaxf(cm, __shfl_xor(cm, off, 32));
            const float nm = fmaxf(rmax[v], cm);
            const float alpha = __expf(rmax[v] - nm);
            rmax[v] = nm;
            const float p0 = __expf(s[0][v] - nm);
            const float p1 = __expf(s[1][v] - nm);
            pch[0][v] = p0; pch[1][v] = p1;
            float ps = p0 + p1;
            #pragma unroll
            for (int off = 1; off < 16; off <<= 1)
                ps += __shfl_xor(ps, off, 32);
            rsum[v] = rsum[v] * alpha + ps;
            #pragma unroll
            for (int nt = 0; nt < 4; nt++) oacc[nt][v] *= alpha;
        }

        // ---- reswizzle P (C layout) -> A-fragment layout via LDS
        __syncthreads();
        #pragma unroll
        for (int st = 0; st < 2; st++)
            #pragma unroll
            for (int v = 0; v < 8; v++)
                Pl[(v + 8 * lh) * 32 + st * 16 + lm] = (half_t)pch[st][v];
        __syncthreads();
        HF16 pf;
        {
            const half_t* pp = Pl + lm * 32 + 8 * lh;
            pf.q[0] = *(const uint4*)pp;
            pf.q[1] = *(const uint4*)(pp + 16);
        }
        #pragma unroll
        for (int nt = 0; nt < 4; nt++) {
            HF16 vf;
            const half_t* vp = Vt + (size_t)(nt * 16 + lm) * T_ + b * SP + t0 + 8 * lh;
            vf.q[0] = *(const uint4*)vp;
            vf.q[1] = *(const uint4*)(vp + 16);
            oacc[nt] = wmma_f16(pf, vf, oacc[nt]);
        }
        __syncthreads();
    }

    // ---- normalize, gelu, store
    #pragma unroll
    for (int nt = 0; nt < 4; nt++) {
        #pragma unroll
        for (int v = 0; v < 8; v++) {
            const float val = oacc[nt][v] / rsum[v];
            const int row = b * SP + mt * 16 + v + 8 * lh;
            G[(size_t)row * E_ + h * 64 + nt * 16 + lm] = (half_t)gelu_exact(val);
        }
    }
}

// ---------------------------------------------------------------------------
// Output: out[b,o] = dot(X[ae[b], sid[o], :], OW[o, :, ad[b]])
// grid = 1024, block = 256 (8 waves; one wave per output element)
// ---------------------------------------------------------------------------
__global__ __launch_bounds__(256) void out_kernel(
    const float* __restrict__ Xf, const float* __restrict__ OW,
    const int* __restrict__ aenv, const int* __restrict__ adoor,
    const int* __restrict__ orids, float* __restrict__ out)
{
    const int wave = threadIdx.x >> 5, lane = threadIdx.x & 31;
    const int oidx = blockIdx.x * 8 + wave;   // 0..8191
    const int b = oidx >> 8, o = oidx & 255;
    const int ae = aenv[b], ad = adoor[b];
    const int sid = orids[o];
    const float* xr = Xf + (size_t)(ae * SP + sid) * E_;
    const float* wr = OW + (size_t)o * E_ * 512 + ad;
    float acc = 0.0f;
    for (int e = lane; e < E_; e += 32)
        acc += xr[e] * wr[(size_t)e * 512];
    #pragma unroll
    for (int off = 16; off > 0; off >>= 1)
        acc += __shfl_xor(acc, off, 32);
    if (lane == 0) out[b * 256 + o] = acc;
}

// ---------------------------------------------------------------------------
// Host launch
// ---------------------------------------------------------------------------
extern "C" void kernel_launch(void* const* d_in, const int* in_sizes, int n_in,
                              void* d_out, int out_size, void* d_ws, size_t ws_size,
                              hipStream_t stream)
{
    (void)in_sizes; (void)n_in; (void)out_size; (void)ws_size;

    const unsigned char* room_mask = (const unsigned char*)d_in[0];
    const int*   rpx        = (const int*)d_in[1];
    const int*   rpy        = (const int*)d_in[2];
    const int*   mdid       = (const int*)d_in[3];
    const int*   aenv       = (const int*)d_in[4];
    const int*   adoor      = (const int*)d_in[5];
    const float* steps      = (const float*)d_in[6];
    const float* temp       = (const float*)d_in[7];
    const float* mc         = (const float*)d_in[8];
    const int*   room_dir   = (const int*)d_in[9];
    const int*   rhx        = (const int*)d_in[10];
    const int*   rhy        = (const int*)d_in[11];
    const int*   orids      = (const int*)d_in[12];
    const float* Wg         = (const float*)d_in[13];
    const float* bg         = (const float*)d_in[14];
    const float* pex        = (const float*)d_in[15];
    const float* pey        = (const float*)d_in[16];
    const float* rem        = (const float*)d_in[17];
    const float* unp        = (const float*)d_in[18];
    const float* mde        = (const float*)d_in[19];
    const float* mpx        = (const float*)d_in[20];
    const float* mpy        = (const float*)d_in[21];
    const float* Wq         = (const float*)d_in[22];
    const float* Wk         = (const float*)d_in[23];
    const float* Wv         = (const float*)d_in[24];
    const float* Wpost      = (const float*)d_in[25];
    const float* W1         = (const float*)d_in[26];
    const float* W2         = (const float*)d_in[27];
    const float* OW         = (const float*)d_in[28];

    // ---- workspace carve-up
    char* ws = (char*)d_ws;
    size_t off = 0;
    auto alloc = [&](size_t bytes) -> void* {
        void* p = ws + off;
        off += (bytes + 255) & ~(size_t)255;
        return p;
    };
    float*  Xf  = (float*) alloc((size_t)T_ * E_ * 4);
    half_t* Xh  = (half_t*)alloc((size_t)T_ * E_ * 2);
    half_t* Qh  = (half_t*)alloc((size_t)T_ * E_ * 2);
    half_t* Kh  = (half_t*)alloc((size_t)T_ * 64 * 2);
    half_t* Vh  = (half_t*)alloc((size_t)T_ * 64 * 2);
    half_t* Vt  = (half_t*)alloc((size_t)64 * T_ * 2);
    half_t* Gh  = (half_t*)alloc((size_t)T_ * E_ * 2);
    half_t* Hh  = (half_t*)alloc((size_t)T_ * HID_ * 2);
    half_t* WqH = (half_t*)alloc((size_t)L_ * E_ * E_ * 2);
    half_t* WkH = (half_t*)alloc((size_t)L_ * E_ * 64 * 2);
    half_t* WvH = (half_t*)alloc((size_t)L_ * E_ * 64 * 2);
    half_t* WpH = (half_t*)alloc((size_t)L_ * E_ * E_ * 2);
    half_t* W1H = (half_t*)alloc((size_t)L_ * E_ * HID_ * 2);
    half_t* W2H = (half_t*)alloc((size_t)L_ * HID_ * E_ * 2);

    // ---- weight conversion f32 -> f16
    auto conv = [&](const float* s, half_t* d, int n) {
        f32_to_f16_kernel<<<(n + 255) / 256, 256, 0, stream>>>(s, d, n);
    };
    conv(Wq,    WqH, L_ * E_ * E_);
    conv(Wk,    WkH, L_ * E_ * 64);
    conv(Wv,    WvH, L_ * E_ * 64);
    conv(Wpost, WpH, L_ * E_ * E_);
    conv(W1,    W1H, L_ * E_ * HID_);
    conv(W2,    W2H, L_ * HID_ * E_);

    // ---- embeddings -> X
    embed_kernel<<<T_, 128, 0, stream>>>(room_mask, rpx, rpy, mdid, steps, temp,
                                         mc, room_dir, rhx, rhy, Wg, bg, pex,
                                         pey, rem, unp, mde, mpx, mpy, Xf, Xh);

    const dim3 blk(256);
    for (int l = 0; l < L_; l++) {
        // Q/K/V projections
        gemm_kernel<0, 128><<<dim3(E_ / 128, T_ / 128), blk, 0, stream>>>(
            Xh, WqH + (size_t)l * E_ * E_, E_, E_, Qh, nullptr);
        gemm_kernel<0, 64><<<dim3(1, T_ / 128), blk, 0, stream>>>(
            Xh, WkH + (size_t)l * E_ * 64, E_, 64, Kh, nullptr);
        gemm_kernel<0, 64><<<dim3(1, T_ / 128), blk, 0, stream>>>(
            Xh, WvH + (size_t)l * E_ * 64, E_, 64, Vh, nullptr);
        transpose_v_kernel<<<(T_ * 64 + 255) / 256, 256, 0, stream>>>(Vh, Vt);
        // attention + fused gelu
        attn_kernel<<<B_ * H_ * (SP / 16), 32, 0, stream>>>(Qh, Kh, Vt, Gh);
        // post projection + residual
        gemm_kernel<2, 128><<<dim3(E_ / 128, T_ / 128), blk, 0, stream>>>(
            Gh, WpH + (size_t)l * E_ * E_, E_, E_, Xh, Xf);
        // FFN
        gemm_kernel<1, 128><<<dim3(HID_ / 128, T_ / 128), blk, 0, stream>>>(
            Xh, W1H + (size_t)l * E_ * HID_, E_, HID_, Hh, nullptr);
        gemm_kernel<2, 128><<<dim3(E_ / 128, T_ / 128), blk, 0, stream>>>(
            Hh, W2H + (size_t)l * HID_ * E_, HID_, E_, Xh, Xf);
    }

    // ---- output gather-dot (only the selected door column per env)
    out_kernel<<<1024, 256, 0, stream>>>(Xf, OW, aenv, adoor, orids,
                                         (float*)d_out);
}
